// ISMBlock_85598698209297
// MI455X (gfx1250) — matrix operand order repaired and checked by reading, not compiled, not run on hardware
//
#include <hip/hip_runtime.h>
#include <hip/hip_bf16.h>
#include <math.h>

// ---------------- types / constants ----------------
typedef _Float16 half8  __attribute__((ext_vector_type(8)));
typedef _Float16 half16 __attribute__((ext_vector_type(16)));
typedef float    float8 __attribute__((ext_vector_type(8)));

#define kB   2
#define kL   1024
#define kD   768
#define kH   8
#define kHD  96
#define kDI  1536
#define kNS  16
#define kKW  4
#define kDTR 48
#define kFF  3072
#define kCG  256
#define kNT  (kB*kL)
#define EPSV 1e-5f

#define TILE_M 64
#define TILE_N 64
#define TILE_K 64
#define LDSR   72   // halves per LDS row (144B): keeps all 8-half groups 16B aligned

__device__ __forceinline__ float gelu_exact(float v){ return 0.5f*v*(1.f+erff(v*0.70710678118654752f)); }
__device__ __forceinline__ float silu_f(float v){ return v/(1.f+expf(-v)); }
__device__ __forceinline__ float softplus_f(float v){ return (v>20.f)?v:log1pf(expf(v)); }

__device__ __forceinline__ half8 pack8(float4 a, float4 b){
    half8 h;
    h[0]=(_Float16)a.x; h[1]=(_Float16)a.y; h[2]=(_Float16)a.z; h[3]=(_Float16)a.w;
    h[4]=(_Float16)b.x; h[5]=(_Float16)b.y; h[6]=(_Float16)b.z; h[7]=(_Float16)b.w;
    return h;
}

// ---------------- generic WMMA GEMM ----------------
// C[m][n] = act(alpha * sum_k A[m][k]*B'[k][n] + bias[n]) (+ resid[m][n])
// BTRANS==0: B'[k][n] = Bm[n*ldb+k]  (weight (N,K) row-major, i.e. A @ W^T)
// BTRANS==1: B'[k][n] = Bm[k*ldb+n]  (plain (K,N) operand, e.g. V in attention)
template<typename BT, int BTRANS, int ACT>
__global__ void __launch_bounds__(128)
gemm_wmma(const float* __restrict__ A, int lda,
          const BT* __restrict__ Bm, int ldb,
          const float* __restrict__ bias,
          const float* __restrict__ resid,
          float* __restrict__ C, int ldc,
          int M, int N, int K, float alpha)
{
    __shared__ __align__(16) _Float16 As[TILE_M*LDSR];
    __shared__ __align__(16) _Float16 Bs[TILE_N*LDSR];
    const int tid  = threadIdx.x;
    const int lane = tid & 31;
    const int wid  = tid >> 5;
    const int mBase = blockIdx.y*TILE_M, nBase = blockIdx.x*TILE_N;
    const int waveM = (wid>>1)*32, waveN = (wid&1)*32;

    float8 acc[2][2] = {};

    const int lrow = tid & 63;          // tile row this thread loads
    const int lcol = (tid >> 6) * 32;   // 32-wide column segment (0 or 32)

    const bool mFull = (mBase + TILE_M <= M);
    const bool nFull = (nBase + TILE_N <= N);
    const bool mOk   = (mBase + lrow) < M;
    const bool nOk   = (nBase + lrow) < N;
    const int  gmC   = min(mBase + lrow, M - 1);   // clamped row indices
    const int  gnC   = min(nBase + lrow, N - 1);

    for (int k0 = 0; k0 < K; k0 += TILE_K) {
        const bool kFull = (k0 + TILE_K <= K);
        // ---- A tile: 64x64 f32 -> f16 in LDS ----
        if (mFull & kFull) {
            const float* ap = A + (size_t)(mBase + lrow)*lda + k0 + lcol;
            _Float16* dst = &As[lrow*LDSR + lcol];
            #pragma unroll
            for (int c = 0; c < 32; c += 8) {
                float4 v0 = *(const float4*)(ap + c);
                float4 v1 = *(const float4*)(ap + c + 4);
                *(half8*)(dst + c) = pack8(v0, v1);
            }
        } else {
            // clamped addresses + select-zero: no exec-mask branching, loads pipeline freely
            const float* ap = A + (size_t)gmC*lda;
            #pragma unroll
            for (int c = 0; c < 32; ++c) {
                const int gk = k0 + lcol + c;
                float v = ap[min(gk, K - 1)];
                As[lrow*LDSR + lcol + c] = (mOk && gk < K) ? (_Float16)v : (_Float16)0.f;
            }
        }
        // ---- B tile: 64(N) x 64(K) -> f16 in LDS (Bs[n][k]) ----
        if (nFull & kFull) {
            if (BTRANS) {
                const BT* bp = Bm + (size_t)(k0 + lcol)*ldb + nBase + lrow;
                _Float16* dst = &Bs[lrow*LDSR + lcol];
                #pragma unroll
                for (int c = 0; c < 32; c += 8) {
                    half8 h;
                    #pragma unroll
                    for (int e = 0; e < 8; ++e)
                        h[e] = (_Float16)(float)bp[(size_t)(c+e)*ldb];
                    *(half8*)(dst + c) = h;
                }
            } else if (sizeof(BT) == 2) {
                const BT* bp = Bm + (size_t)(nBase + lrow)*ldb + k0 + lcol;
                _Float16* dst = &Bs[lrow*LDSR + lcol];
                #pragma unroll
                for (int c = 0; c < 32; c += 8)
                    *(half8*)(dst + c) = *(const half8*)(bp + c);
            } else {
                const BT* bp = Bm + (size_t)(nBase + lrow)*ldb + k0 + lcol;
                _Float16* dst = &Bs[lrow*LDSR + lcol];
                #pragma unroll
                for (int c = 0; c < 32; c += 8) {
                    float4 v0 = *(const float4*)((const float*)bp + c);
                    float4 v1 = *(const float4*)((const float*)bp + c + 4);
                    *(half8*)(dst + c) = pack8(v0, v1);
                }
            }
        } else {
            #pragma unroll
            for (int c = 0; c < 32; ++c) {
                const int gk  = k0 + lcol + c;
                const int gkC = min(gk, K - 1);
                float v = BTRANS ? (float)Bm[(size_t)gkC*ldb + gnC]
                                 : (float)Bm[(size_t)gnC*ldb + gkC];
                Bs[lrow*LDSR + lcol + c] = (nOk && gk < K) ? (_Float16)v : (_Float16)0.f;
            }
        }
        __syncthreads();

        // ---- fragments + 8 WMMA per tile (ISA 7.12.2 layouts) ----
        const int am  = lane & 15;
        const int akb = (lane < 16) ? 0 : 8;    // A: half-wave K base
        const int bn  = lane & 15;
        const int bkb = (lane < 16) ? 0 : 8;    // B: half-wave K base (same order as A)
        #pragma unroll
        for (int kk = 0; kk < TILE_K; kk += 32) {
            union F { half16 v; half8 h[2]; } aF[2], bF[2];
            #pragma unroll
            for (int i = 0; i < 2; ++i) {
                const _Float16* p = &As[(waveM + i*16 + am)*LDSR + kk];
                aF[i].h[0] = *(const half8*)(p + akb);
                aF[i].h[1] = *(const half8*)(p + 16 + akb);
            }
            #pragma unroll
            for (int j = 0; j < 2; ++j) {
                const _Float16* p = &Bs[(waveN + j*16 + bn)*LDSR + kk];
                bF[j].h[0] = *(const half8*)(p + bkb);
                bF[j].h[1] = *(const half8*)(p + 16 + bkb);
            }
            #pragma unroll
            for (int i = 0; i < 2; ++i)
                #pragma unroll
                for (int j = 0; j < 2; ++j)
                    acc[i][j] = __builtin_amdgcn_wmma_f32_16x16x32_f16(
                        false, aF[i].v, false, bF[j].v, (short)0, acc[i][j], false, false);
        }
        __syncthreads();
    }

    // epilogue: C layout — lane<16: N=lane, M=r; lane>=16: N=lane-16, M=r+8
    #pragma unroll
    for (int i = 0; i < 2; ++i)
        #pragma unroll
        for (int j = 0; j < 2; ++j) {
            const int n = nBase + waveN + j*16 + (lane & 15);
            if (n >= N) continue;
            const float bv = bias ? bias[n] : 0.f;
            #pragma unroll
            for (int r = 0; r < 8; ++r) {
                const int m = mBase + waveM + i*16 + ((lane < 16) ? r : r + 8);
                if (m >= M) continue;
                float v = acc[i][j][r]*alpha + bv;
                if      (ACT == 1) v = gelu_exact(v);
                else if (ACT == 2) v = silu_f(v);
                else if (ACT == 3) v = softplus_f(v);
                if (resid) v += resid[(size_t)m*ldc + n];
                C[(size_t)m*ldc + n] = v;
            }
        }
}

// ---------------- elementwise / reduction kernels ----------------
__global__ void cvt_f32_f16(const float* __restrict__ s, _Float16* __restrict__ d, int n){
    int i = blockIdx.x*256 + threadIdx.x;
    if (i < n) d[i] = (_Float16)s[i];
}

__global__ void softmax_rows(float* __restrict__ S, int N){
    float* p = S + (size_t)blockIdx.x * N;
    __shared__ float red[256];
    const int tid = threadIdx.x;
    float mx = -1e30f;
    for (int i = tid; i < N; i += 256) mx = fmaxf(mx, p[i]);
    red[tid] = mx; __syncthreads();
    for (int s = 128; s > 0; s >>= 1){ if (tid < s) red[tid] = fmaxf(red[tid], red[tid+s]); __syncthreads(); }
    mx = red[0]; __syncthreads();
    float sm = 0.f;
    for (int i = tid; i < N; i += 256){ float e = expf(p[i]-mx); p[i] = e; sm += e; }
    red[tid] = sm; __syncthreads();
    for (int s = 128; s > 0; s >>= 1){ if (tid < s) red[tid] += red[tid+s]; __syncthreads(); }
    const float inv = 1.f/red[0];
    for (int i = tid; i < N; i += 256) p[i] *= inv;
}

// out = LN(in1 (+ in2)) * g + b, rows of width kD
__global__ void layernorm_row(const float* __restrict__ in1, const float* __restrict__ in2,
                              const float* __restrict__ g, const float* __restrict__ b,
                              float* __restrict__ out){
    const size_t row = blockIdx.x;
    const float* p1 = in1 + row*kD;
    const float* p2 = in2 ? in2 + row*kD : nullptr;
    __shared__ float r1[256], r2[256];
    const int tid = threadIdx.x;
    float s = 0.f, sq = 0.f;
    for (int i = tid; i < kD; i += 256){
        float v = p1[i] + (p2 ? p2[i] : 0.f);
        s += v; sq += v*v;
    }
    r1[tid] = s; r2[tid] = sq; __syncthreads();
    for (int st = 128; st > 0; st >>= 1){ if (tid < st){ r1[tid]+=r1[tid+st]; r2[tid]+=r2[tid+st]; } __syncthreads(); }
    const float mean = r1[0]/kD;
    const float var  = r2[0]/kD - mean*mean;
    const float inv  = rsqrtf(var + EPSV);
    for (int i = tid; i < kD; i += 256){
        float v = p1[i] + (p2 ? p2[i] : 0.f);
        out[row*kD + i] = (v - mean)*inv*g[i] + b[i];
    }
}

// multi-scale grouped conv branch + exact GELU -> (B,L,768) row-major
__global__ void glce_conv(const float* __restrict__ x,
                          const float* __restrict__ w3, const float* __restrict__ b3,
                          const float* __restrict__ w5, const float* __restrict__ b5,
                          const float* __restrict__ w7, const float* __restrict__ b7,
                          float* __restrict__ out){
    int idx = blockIdx.x*256 + threadIdx.x;
    if (idx >= kNT*kD) return;
    const int c = idx % kD;
    const int l = (idx / kD) % kL;
    const int bb = idx / (kD*kL);
    const int kind = c / kCG, cg = c % kCG;
    const float* w; const float* bs; int wk, pad;
    if (kind == 0){ w = w3; bs = b3; wk = 3; pad = 1; }
    else if (kind == 1){ w = w5; bs = b5; wk = 5; pad = 2; }
    else { w = w7; bs = b7; wk = 7; pad = 3; }
    float acc = bs[cg];
    #pragma unroll
    for (int i = 0; i < 3; ++i){
        const int ch = cg*3 + i;
        for (int t = 0; t < wk; ++t){
            const int ll = l - pad + t;
            if (ll >= 0 && ll < kL)
                acc += w[(cg*3 + i)*wk + t] * x[((size_t)bb*kL + ll)*kD + ch];
        }
    }
    out[idx] = gelu_exact(acc);
}

// causal depthwise conv (K=4) over xi (first half of xi_z, stride 2*DI) + SiLU
__global__ void mamba_conv(const float* __restrict__ xiz, const float* __restrict__ cw,
                           const float* __restrict__ cb, float* __restrict__ xc){
    int idx = blockIdx.x*256 + threadIdx.x;
    if (idx >= kNT*kDI) return;
    const int d = idx % kDI;
    const int l = (idx / kDI) % kL;
    const int bb = idx / (kDI*kL);
    float acc = cb[d];
    #pragma unroll
    for (int t = 0; t < kKW; ++t){
        const int ll = l - (kKW-1) + t;
        if (ll >= 0) acc += cw[d*kKW + t] * xiz[((size_t)bb*kL + ll)*(2*kDI) + d];
    }
    xc[idx] = silu_f(acc);
}

// selective scan: one thread per channel d, N=16 states in registers.
// prod[b,l,d] = (scan_y + xc*D) * silu(z)
__global__ void __launch_bounds__(256)
mamba_scan(const float* __restrict__ delta, const float* __restrict__ xc,
           const float* __restrict__ dbc,   const float* __restrict__ xiz,
           const float* __restrict__ A_log, const float* __restrict__ Dp,
           float* __restrict__ prod){
    const int bb = blockIdx.x;
    const int d  = blockIdx.y*256 + threadIdx.x;   // 6 blocks * 256 = 1536
    float a[kNS], h[kNS];
    #pragma unroll
    for (int n = 0; n < kNS; ++n){ a[n] = -expf(A_log[d*kNS + n]); h[n] = 0.f; }
    const float Dv = Dp[d];
    __shared__ float sB[kNS], sC[kNS];
    for (int l = 0; l < kL; ++l){
        const size_t row = (size_t)bb*kL + l;
        if (threadIdx.x < kNS)            sB[threadIdx.x]      = dbc[row*80 + kDTR + threadIdx.x];
        else if (threadIdx.x < 2*kNS)     sC[threadIdx.x-kNS]  = dbc[row*80 + kDTR + kNS + (threadIdx.x-kNS)];
        __syncthreads();
        const float dl = delta[row*kDI + d];
        const float u  = xc[row*kDI + d];
        float accy = 0.f;
        #pragma unroll
        for (int n = 0; n < kNS; ++n){
            const float dA = expf(dl * a[n]);
            h[n] = dA*h[n] + dl*sB[n]*u;
            accy += h[n]*sC[n];
        }
        const float y = accy + u*Dv;
        const float z = xiz[row*(2*kDI) + kDI + d];
        prod[row*kDI + d] = y * silu_f(z);
        __syncthreads();
    }
}

__global__ void flip_rows(const float* __restrict__ in, float* __restrict__ out){
    int idx = blockIdx.x*256 + threadIdx.x;
    if (idx >= kNT*kD) return;
    const int c = idx % kD;
    const int l = (idx / kD) % kL;
    const int bb = idx / (kD*kL);
    out[idx] = in[((size_t)bb*kL + (kL-1-l))*kD + c];
}

__global__ void combine_x2(const float* __restrict__ x1, const float* __restrict__ yf,
                           const float* __restrict__ yb, float* __restrict__ x2){
    int idx = blockIdx.x*256 + threadIdx.x;
    if (idx >= kNT*kD) return;
    const int c = idx % kD;
    const int l = (idx / kD) % kL;
    const int bb = idx / (kD*kL);
    x2[idx] = x1[idx] + yf[idx] + yb[((size_t)bb*kL + (kL-1-l))*kD + c];
}

__global__ void ew_mul(float* __restrict__ a, const float* __restrict__ b, int n){
    int i = blockIdx.x*256 + threadIdx.x;
    if (i < n) a[i] *= b[i];
}

// ---------------- host-side launch helpers ----------------
static void gemm16(hipStream_t st, const float* A, int lda, const _Float16* Bw, int ldb,
                   const float* bias, const float* resid, float* C, int ldc,
                   int M, int N, int K, float alpha, int act){
    dim3 g((N+TILE_N-1)/TILE_N, (M+TILE_M-1)/TILE_M);
    if      (act == 0) gemm_wmma<_Float16,0,0><<<g,128,0,st>>>(A,lda,Bw,ldb,bias,resid,C,ldc,M,N,K,alpha);
    else if (act == 1) gemm_wmma<_Float16,0,1><<<g,128,0,st>>>(A,lda,Bw,ldb,bias,resid,C,ldc,M,N,K,alpha);
    else if (act == 2) gemm_wmma<_Float16,0,2><<<g,128,0,st>>>(A,lda,Bw,ldb,bias,resid,C,ldc,M,N,K,alpha);
    else               gemm_wmma<_Float16,0,3><<<g,128,0,st>>>(A,lda,Bw,ldb,bias,resid,C,ldc,M,N,K,alpha);
}

extern "C" void kernel_launch(void* const* d_in, const int* in_sizes, int n_in,
                              void* d_out, int out_size, void* d_ws, size_t ws_size,
                              hipStream_t stream) {
    // ---- inputs (setup_inputs order) ----
    const float* x        = (const float*)d_in[0];
    const float* qkv_w    = (const float*)d_in[1];
    const float* qkv_b    = (const float*)d_in[2];
    const float* attout_w = (const float*)d_in[3];
    const float* attout_b = (const float*)d_in[4];
    const float* conv3_w  = (const float*)d_in[5];
    const float* conv3_b  = (const float*)d_in[6];
    const float* conv5_w  = (const float*)d_in[7];
    const float* conv5_b  = (const float*)d_in[8];
    const float* conv7_w  = (const float*)d_in[9];
    const float* conv7_b  = (const float*)d_in[10];
    const float* gproj_w  = (const float*)d_in[11];
    const float* gproj_b  = (const float*)d_in[12];
    const float* lproj_w  = (const float*)d_in[13];
    const float* lproj_b  = (const float*)d_in[14];
    const float* fus_w    = (const float*)d_in[15];
    const float* fus_b    = (const float*)d_in[16];
    const float* glce_g   = (const float*)d_in[17];
    const float* glce_bb  = (const float*)d_in[18];
    const float* ssm_g    = (const float*)d_in[19];
    const float* ssm_bb   = (const float*)d_in[20];
    const float* din_w[2]  = {(const float*)d_in[21], (const float*)d_in[30]};
    const float* dconv_w[2]= {(const float*)d_in[22], (const float*)d_in[31]};
    const float* dconv_b[2]= {(const float*)d_in[23], (const float*)d_in[32]};
    const float* dxp_w[2]  = {(const float*)d_in[24], (const float*)d_in[33]};
    const float* ddt_w[2]  = {(const float*)d_in[25], (const float*)d_in[34]};
    const float* ddt_b[2]  = {(const float*)d_in[26], (const float*)d_in[35]};
    const float* dA_log[2] = {(const float*)d_in[27], (const float*)d_in[36]};
    const float* dD[2]     = {(const float*)d_in[28], (const float*)d_in[37]};
    const float* dout_w[2] = {(const float*)d_in[29], (const float*)d_in[38]};
    const float* ffn_g    = (const float*)d_in[39];
    const float* ffn_bb   = (const float*)d_in[40];
    const float* gate_w   = (const float*)d_in[41];
    const float* up_w     = (const float*)d_in[42];
    const float* down_w   = (const float*)d_in[43];
    float* out = (float*)d_out;

    // ---- workspace layout (f32 arena then f16 weight arena) ----
    float* W = (float*)d_ws;
    size_t off = 0;
    auto fa = [&](size_t n){ float* p = W + off; off += n; return p; };
    float* qkvB  = fa((size_t)kNT*2304);
    float* Sb    = fa((size_t)kL*kL);
    float* oB    = fa((size_t)kNT*kD);
    float* attn  = fa((size_t)kNT*kD);
    float* cat   = fa((size_t)kNT*kD);
    float* locp  = fa((size_t)kNT*kD);
    float* fused = fa((size_t)kNT*kD);
    float* x1    = fa((size_t)kNT*kD);
    float* xn    = fa((size_t)kNT*kD);
    float* xnf   = fa((size_t)kNT*kD);
    float* x2    = fa((size_t)kNT*kD);
    float* xn2   = fa((size_t)kNT*kD);
    float* ytF   = fa((size_t)kNT*kD);
    float* ytB   = fa((size_t)kNT*kD);
    float* xiz   = fa((size_t)kNT*2*kDI);
    float* xc    = fa((size_t)kNT*kDI);
    float* dbcB  = fa((size_t)kNT*80);
    float* delta = fa((size_t)kNT*kDI);
    float* prod  = fa((size_t)kNT*kDI);
    float* gateB = qkvB;   // overlay: dead attention buffers (6.3M <= qkv+S+o)
    float* upB   = xiz;    // overlay: dead mamba buffer (exactly 6.3M)

    off = (off + 7) & ~(size_t)7;
    _Float16* Harena = (_Float16*)(W + off);
    size_t hoff = 0;
    auto ha = [&](size_t n){ _Float16* p = Harena + hoff; hoff += n; return p; };
    _Float16* qkvW16  = ha((size_t)2304*kD);
    _Float16* aoW16   = ha((size_t)kD*kD);
    _Float16* gpW16   = ha((size_t)(kD/2)*kD);
    _Float16* lpW16   = ha((size_t)(kD/2)*kD);
    _Float16* fusW16  = ha((size_t)kD*kD);
    _Float16* inW16[2]  = { ha((size_t)2*kDI*kD), ha((size_t)2*kDI*kD) };
    _Float16* xpW16[2]  = { ha((size_t)80*kDI),   ha((size_t)80*kDI)   };
    _Float16* dtW16[2]  = { ha((size_t)kDI*kDTR), ha((size_t)kDI*kDTR) };
    _Float16* outW16[2] = { ha((size_t)kD*kDI),   ha((size_t)kD*kDI)   };
    _Float16* gateW16 = ha((size_t)kFF*kD);
    _Float16* upW16   = ha((size_t)kFF*kD);
    _Float16* downW16 = ha((size_t)kD*kFF);

    auto cvt = [&](const float* s, _Float16* d, size_t n){
        cvt_f32_f16<<<(unsigned)((n+255)/256),256,0,stream>>>(s,d,(int)n);
    };
    cvt(qkv_w,  qkvW16, (size_t)2304*kD);
    cvt(attout_w, aoW16, (size_t)kD*kD);
    cvt(gproj_w, gpW16, (size_t)(kD/2)*kD);
    cvt(lproj_w, lpW16, (size_t)(kD/2)*kD);
    cvt(fus_w,  fusW16, (size_t)kD*kD);
    for (int dir = 0; dir < 2; ++dir){
        cvt(din_w[dir],  inW16[dir],  (size_t)2*kDI*kD);
        cvt(dxp_w[dir],  xpW16[dir],  (size_t)80*kDI);
        cvt(ddt_w[dir],  dtW16[dir],  (size_t)kDI*kDTR);
        cvt(dout_w[dir], outW16[dir], (size_t)kD*kDI);
    }
    cvt(gate_w, gateW16, (size_t)kFF*kD);
    cvt(up_w,   upW16,   (size_t)kFF*kD);
    cvt(down_w, downW16, (size_t)kD*kFF);

    // ---- 1) MHA ----
    gemm16(stream, x, kD, qkvW16, kD, qkv_b, nullptr, qkvB, 2304, kNT, 2304, kD, 1.f, 0);
    const float scal = 0.102062072615966f; // 1/sqrt(96)
    for (int bb = 0; bb < kB; ++bb)
        for (int h = 0; h < kH; ++h){
            const float* Q  = qkvB + (size_t)bb*kL*2304 + h*kHD;
            const float* Kp = qkvB + (size_t)bb*kL*2304 + kD + h*kHD;
            const float* V  = qkvB + (size_t)bb*kL*2304 + 2*kD + h*kHD;
            gemm_wmma<float,0,0><<<dim3(kL/64,kL/64),128,0,stream>>>(
                Q, 2304, Kp, 2304, nullptr, nullptr, Sb, kL, kL, kL, kHD, scal);
            softmax_rows<<<kL,256,0,stream>>>(Sb, kL);
            gemm_wmma<float,1,0><<<dim3(2,kL/64),128,0,stream>>>(
                Sb, kL, V, 2304, nullptr, nullptr,
                oB + (size_t)bb*kL*kD + h*kHD, kD, kL, kHD, kL, 1.f);
        }
    gemm16(stream, oB, kD, aoW16, kD, attout_b, nullptr, attn, kD, kNT, kD, kD, 1.f, 0);

    // ---- 2) GLCE branch: g | loc -> fuse ----
    gemm16(stream, attn, kD, gpW16, kD, gproj_b, nullptr, cat, kD, kNT, kD/2, kD, 1.f, 0);
    glce_conv<<<(kNT*kD+255)/256,256,0,stream>>>(x, conv3_w,conv3_b, conv5_w,conv5_b, conv7_w,conv7_b, locp);
    gemm16(stream, locp, kD, lpW16, kD, lproj_b, nullptr, cat + kD/2, kD, kNT, kD/2, kD, 1.f, 0);
    gemm16(stream, cat, kD, fusW16, kD, fus_b, nullptr, fused, kD, kNT, kD, kD, 1.f, 0);
    layernorm_row<<<kNT,256,0,stream>>>(x, fused, glce_g, glce_bb, x1);
    layernorm_row<<<kNT,256,0,stream>>>(x1, nullptr, ssm_g, ssm_bb, xn);

    // ---- 3) bidirectional Mamba ----
    flip_rows<<<(kNT*kD+255)/256,256,0,stream>>>(xn, xnf);
    for (int dir = 0; dir < 2; ++dir){
        const float* Xd = (dir == 0) ? xn : xnf;
        float* yout     = (dir == 0) ? ytF : ytB;
        gemm16(stream, Xd, kD, inW16[dir], kD, nullptr, nullptr, xiz, 2*kDI, kNT, 2*kDI, kD, 1.f, 0);
        mamba_conv<<<(kNT*kDI+255)/256,256,0,stream>>>(xiz, dconv_w[dir], dconv_b[dir], xc);
        gemm16(stream, xc, kDI, xpW16[dir], kDI, nullptr, nullptr, dbcB, 80, kNT, 80, kDI, 1.f, 0);
        gemm16(stream, dbcB, 80, dtW16[dir], kDTR, ddt_b[dir], nullptr, delta, kDI, kNT, kDI, kDTR, 1.f, 3);
        mamba_scan<<<dim3(kB, kDI/256),256,0,stream>>>(delta, xc, dbcB, xiz, dA_log[dir], dD[dir], prod);
        gemm16(stream, prod, kDI, outW16[dir], kDI, nullptr, nullptr, yout, kD, kNT, kD, kDI, 1.f, 0);
    }
    combine_x2<<<(kNT*kD+255)/256,256,0,stream>>>(x1, ytF, ytB, x2);

    // ---- 4) SwiGLU FFN with fused residual to d_out ----
    layernorm_row<<<kNT,256,0,stream>>>(x2, nullptr, ffn_g, ffn_bb, xn2);
    gemm16(stream, xn2, kD, gateW16, kD, nullptr, nullptr, gateB, kFF, kNT, kFF, kD, 1.f, 2); // SiLU fused
    gemm16(stream, xn2, kD, upW16,   kD, nullptr, nullptr, upB,   kFF, kNT, kFF, kD, 1.f, 0);
    ew_mul<<<(kNT*kFF+255)/256,256,0,stream>>>(gateB, upB, kNT*kFF);
    gemm16(stream, gateB, kFF, downW16, kFF, nullptr, x2, out, kD, kNT, kD, kFF, 1.f, 0);
}
